// Speller_58720792871086
// MI455X (gfx1250) — matrix operand order repaired and checked by reading, not compiled, not run on hardware
//
#include <hip/hip_runtime.h>
#include <hip/hip_bf16.h>
#include <math.h>

// ---------------------------------------------------------------------------
// LAS "Speller" decoder for MI455X (gfx1250, wave32, WMMA).
//
// Structure:
//   prep:   fp32 -> bf16 weight conversion (gate matrices), once
//   proj:   key_enc / value projections  (WMMA bf16, 16x16x32, fp32 accum)
//   loop 64 steps (graph-captured launches):
//     build_x -> gates1 (WMMA) -> lstm1 update -> gates2 (WMMA) -> lstm2 update
//     query (VALU) -> attention (masked softmax + context, VALU, L2-resident)
//     logits + argmax "preds"
// ---------------------------------------------------------------------------

typedef __attribute__((ext_vector_type(16))) __bf16 v16bf;
typedef __attribute__((ext_vector_type(8)))  __bf16 v8bf;
typedef __attribute__((ext_vector_type(8)))  float  v8f;
typedef __attribute__((ext_vector_type(4)))  float  v4f;

#define BATCH  32
#define TENC   2048
#define LIS    512
#define HID    512
#define EMB    256
#define CLS    64
#define KEY    128
#define VAL    128
#define TDEC   65
#define TSTEPS 64
#define GATES  2048   // 4*HID
#define XDIM   384    // EMB+VAL

__device__ __forceinline__ v8f wmma_bf16(v16bf a, v16bf b, v8f c) {
  // D = A(16x32 bf16) x B(32x16 bf16) + C(16x16 f32)
  return __builtin_amdgcn_wmma_f32_16x16x32_bf16(false, a, false, b,
                                                 (short)0, c, false, false);
}

// A-fragment (16x32 bf16) for one lane from an LDS row (row-major, bf16).
// Layout: lanes 0-15 hold M=lane, lanes 16-31 hold M=lane-16 with K+16;
// elements 0..15 -> K = kc + 16*khalf + e (contiguous, matches B fragments).
__device__ __forceinline__ v16bf a_frag(const __bf16* rowptr, int kc, int khalf) {
  v8bf lo = *(const v8bf*)(rowptr + kc + 16 * khalf);
  v8bf hi = *(const v8bf*)(rowptr + kc + 16 * khalf + 8);
  v16bf a;
#pragma unroll
  for (int e = 0; e < 8; ++e) { a[e] = lo[e]; a[8 + e] = hi[e]; }
  return a;
}

__device__ __forceinline__ float sigm(float x) { return 1.0f / (1.0f + __expf(-x)); }

// ---------------------------------------------------------------------------
// utility kernels
// ---------------------------------------------------------------------------
__global__ void cvt_bf16_kernel(const float* __restrict__ s, __bf16* __restrict__ d, int n) {
  int i = blockIdx.x * 256 + threadIdx.x;
  if (i < n) d[i] = (__bf16)s[i];
}
__global__ void copy_f32_kernel(const float* __restrict__ s, float* __restrict__ d, int n) {
  int i = blockIdx.x * 256 + threadIdx.x;
  if (i < n) d[i] = s[i];
}
__global__ void zero_f32_kernel(float* __restrict__ d, int n) {
  int i = blockIdx.x * 256 + threadIdx.x;
  if (i < n) d[i] = 0.0f;
}

// ---------------------------------------------------------------------------
// key_enc / value projection: C[65536,128] = X[65536,512] @ W^T, + bias,
// value additionally masked.  One block = 16 rows, 8 waves = 8 col-tiles.
// ---------------------------------------------------------------------------
__global__ void __launch_bounds__(256)
proj_kv_kernel(const float* __restrict__ X, const float* __restrict__ Wh,
               const float* __restrict__ bh, const float* __restrict__ Wv,
               const float* __restrict__ bv, const int* __restrict__ lens,
               float* __restrict__ key_enc, float* __restrict__ value) {
  __shared__ __align__(32) __bf16 As[16][LIS];
  const int rowbase = blockIdx.x * 16;
  for (int idx = threadIdx.x; idx < 16 * LIS; idx += 256) {
    int r = idx >> 9, k = idx & 511;
    As[r][k] = (__bf16)X[(size_t)(rowbase + r) * LIS + k];
  }
  __syncthreads();

  const int wave = threadIdx.x >> 5, lane = threadIdx.x & 31;
  const int n = wave * 16 + (lane & 15);   // output column 0..127
  const int khalf = lane >> 4;
  const int m = lane & 15;

  v8f ck, cv;
#pragma unroll
  for (int r = 0; r < 8; ++r) { ck[r] = bh[n]; cv[r] = bv[n]; }

  const float* whrow = Wh + (size_t)n * LIS;
  const float* wvrow = Wv + (size_t)n * LIS;
  for (int kc = 0; kc < LIS; kc += 32) {
    const int kb = kc + 16 * khalf;
    v16bf bk, bvv;
#pragma unroll
    for (int e = 0; e < 16; e += 4) {
      v4f w1 = *(const v4f*)(whrow + kb + e);
      v4f w2 = *(const v4f*)(wvrow + kb + e);
#pragma unroll
      for (int j = 0; j < 4; ++j) { bk[e + j] = (__bf16)w1[j]; bvv[e + j] = (__bf16)w2[j]; }
    }
    v16bf a = a_frag(&As[m][0], kc, khalf);
    ck = wmma_bf16(a, bk, ck);
    cv = wmma_bf16(a, bvv, cv);
  }

#pragma unroll
  for (int r = 0; r < 8; ++r) {
    const int mr = r + 8 * khalf;          // 0..15
    const int row = rowbase + mr;
    const int b = row >> 11, t = row & 2047;
    const float mk = (b == 0) ? 1.0f : ((t < lens[b]) ? 1.0f : 0.0f);
    key_enc[(size_t)row * KEY + n] = ck[r];
    value[(size_t)row * VAL + n] = cv[r] * mk;
  }
}

// ---------------------------------------------------------------------------
// x = [embed[tok], context]  (tokens from padded_targets; tf == 1)
// ---------------------------------------------------------------------------
__global__ void __launch_bounds__(1024)
build_x_kernel(const int* __restrict__ targets, int step, const float* __restrict__ embed,
               const float* __restrict__ context, float* __restrict__ x) {
  const int col = (step > 0) ? (step - 1) : 0;
  for (int idx = threadIdx.x; idx < BATCH * XDIM; idx += 1024) {
    const int b = idx / XDIM, e = idx - b * XDIM;
    const int tok = targets[b * TDEC + col];
    x[idx] = (e < EMB) ? embed[tok * EMB + e] : context[b * VAL + (e - EMB)];
  }
}

// ---------------------------------------------------------------------------
// gates g[32,2048] = A1[32,K1] @ W1^T + A2[32,K2] @ W2^T + (bih + bhh)
// W1/W2 are bf16, row-major [2048, K].  16 blocks x 8 waves cover N=2048;
// each wave owns a 16-col tile and both 16-row M-tiles.
// ---------------------------------------------------------------------------
__global__ void __launch_bounds__(256)
gates_kernel(const float* __restrict__ A1, int K1, const __bf16* __restrict__ W1,
             const float* __restrict__ A2, int K2, const __bf16* __restrict__ W2,
             const float* __restrict__ bih, const float* __restrict__ bhh,
             float* __restrict__ g) {
  __shared__ __align__(32) __bf16 a1s[BATCH][HID];
  __shared__ __align__(32) __bf16 a2s[BATCH][HID];
  for (int idx = threadIdx.x; idx < BATCH * K1; idx += 256) {
    int b = idx / K1, e = idx - b * K1;
    a1s[b][e] = (__bf16)A1[idx];
  }
  for (int idx = threadIdx.x; idx < BATCH * K2; idx += 256) {
    int b = idx >> 9, e = idx & 511;
    a2s[b][e] = (__bf16)A2[idx];
  }
  __syncthreads();

  const int wave = threadIdx.x >> 5, lane = threadIdx.x & 31;
  const int n = blockIdx.x * 128 + wave * 16 + (lane & 15);
  const int khalf = lane >> 4;
  const int m = lane & 15;

  const float bias = bih[n] + bhh[n];
  v8f c0, c1;
#pragma unroll
  for (int r = 0; r < 8; ++r) { c0[r] = bias; c1[r] = bias; }

  const __bf16* w1row = W1 + (size_t)n * K1;
  for (int kc = 0; kc < K1; kc += 32) {
    v16bf bfrag = *(const v16bf*)(w1row + kc + 16 * khalf);
    v16bf a0 = a_frag(&a1s[m][0], kc, khalf);
    v16bf a1 = a_frag(&a1s[16 + m][0], kc, khalf);
    c0 = wmma_bf16(a0, bfrag, c0);
    c1 = wmma_bf16(a1, bfrag, c1);
  }
  const __bf16* w2row = W2 + (size_t)n * K2;
  for (int kc = 0; kc < K2; kc += 32) {
    v16bf bfrag = *(const v16bf*)(w2row + kc + 16 * khalf);
    v16bf a0 = a_frag(&a2s[m][0], kc, khalf);
    v16bf a1 = a_frag(&a2s[16 + m][0], kc, khalf);
    c0 = wmma_bf16(a0, bfrag, c0);
    c1 = wmma_bf16(a1, bfrag, c1);
  }

#pragma unroll
  for (int r = 0; r < 8; ++r) {
    const int mr = r + 8 * khalf;
    g[(size_t)mr * GATES + n] = c0[r];
    g[(size_t)(16 + mr) * GATES + n] = c1[r];
  }
}

// ---------------------------------------------------------------------------
// LSTM elementwise update (torch gate order i,f,g,o)
// ---------------------------------------------------------------------------
__global__ void __launch_bounds__(1024)
lstm_update_kernel(const float* __restrict__ g, float* __restrict__ h, float* __restrict__ c) {
  const int idx = blockIdx.x * 1024 + threadIdx.x;   // b*512 + j
  if (idx >= BATCH * HID) return;
  const int b = idx >> 9, j = idx & 511;
  const float* gr = g + (size_t)b * GATES;
  const float ig = sigm(gr[j]);
  const float fg = sigm(gr[HID + j]);
  const float gg = tanhf(gr[2 * HID + j]);
  const float og = sigm(gr[3 * HID + j]);
  const float cn = fg * c[idx] + ig * gg;
  c[idx] = cn;
  h[idx] = og * tanhf(cn);
}

// ---------------------------------------------------------------------------
// query[32,128] = h2 @ W_s^T + b_s
// ---------------------------------------------------------------------------
__global__ void __launch_bounds__(1024)
query_kernel(const float* __restrict__ h2, const float* __restrict__ Ws,
             const float* __restrict__ bs, float* __restrict__ q) {
  for (int idx = threadIdx.x; idx < BATCH * KEY; idx += 1024) {
    const int b = idx >> 7, k = idx & 127;
    const float* hp = h2 + (size_t)b * HID;
    const float* wp = Ws + (size_t)k * HID;
    float acc = bs[k];
    for (int j = 0; j < HID; j += 4) {
      v4f hh = *(const v4f*)(hp + j);
      v4f ww = *(const v4f*)(wp + j);
      acc += hh[0] * ww[0] + hh[1] * ww[1] + hh[2] * ww[2] + hh[3] * ww[3];
    }
    q[idx] = acc;
  }
}

// ---------------------------------------------------------------------------
// masked softmax attention + context.  One block per batch row.
// ---------------------------------------------------------------------------
__global__ void __launch_bounds__(1024)
attention_kernel(const float* __restrict__ query, const float* __restrict__ key_enc,
                 const float* __restrict__ value, const int* __restrict__ lens,
                 float* __restrict__ context, float* __restrict__ atts_out, int step) {
  __shared__ float qs[KEY];
  __shared__ float red[1024];
  __shared__ float att[TENC];
  __shared__ float ctxp[8][VAL];
  const int b = blockIdx.x, tid = threadIdx.x;
  if (tid < KEY) qs[tid] = query[b * KEY + tid];
  __syncthreads();
  const int len = lens[b];

  float e[2];
#pragma unroll
  for (int p = 0; p < 2; ++p) {
    const int t = tid + p * 1024;
    const bool mk = (b == 0) || (t < len);
    float s = -1e30f;
    if (mk) {
      const float* kp = key_enc + ((size_t)b * TENC + t) * KEY;
      float acc = 0.0f;
      for (int k = 0; k < KEY; k += 4) {
        v4f kk = *(const v4f*)(kp + k);
        v4f qq = *(const v4f*)(qs + k);
        acc += kk[0] * qq[0] + kk[1] * qq[1] + kk[2] * qq[2] + kk[3] * qq[3];
      }
      s = acc;
    }
    e[p] = s;
  }

  // max reduce
  red[tid] = fmaxf(e[0], e[1]);
  __syncthreads();
  for (int off = 512; off > 0; off >>= 1) {
    if (tid < off) red[tid] = fmaxf(red[tid], red[tid + off]);
    __syncthreads();
  }
  const float mx = red[0];
  __syncthreads();

  float ssum = 0.0f;
#pragma unroll
  for (int p = 0; p < 2; ++p) {
    const int t = tid + p * 1024;
    const float a = (e[p] > -1e29f) ? __expf(e[p] - mx) : 0.0f;
    att[t] = a;
    ssum += a;
  }
  red[tid] = ssum;
  __syncthreads();
  for (int off = 512; off > 0; off >>= 1) {
    if (tid < off) red[tid] += red[tid + off];
    __syncthreads();
  }
  const float inv = 1.0f / fmaxf(red[0], 1e-12f);

#pragma unroll
  for (int p = 0; p < 2; ++p) {
    const int t = tid + p * 1024;
    const float a = att[t] * inv;
    att[t] = a;
    atts_out[((size_t)step * BATCH + b) * TENC + t] = a;
  }
  __syncthreads();

  // context: 8 t-groups x 128 lanes, coalesced over v
  const int gq = tid >> 7, v = tid & 127;
  const float* vp = value + (size_t)b * TENC * VAL + v;
  float part = 0.0f;
  for (int t = gq * 256; t < gq * 256 + 256; ++t) part += att[t] * vp[(size_t)t * VAL];
  ctxp[gq][v] = part;
  __syncthreads();
  if (tid < VAL) {
    float s = 0.0f;
#pragma unroll
    for (int gg = 0; gg < 8; ++gg) s += ctxp[gg][tid];
    context[b * VAL + tid] = s;
  }
}

// ---------------------------------------------------------------------------
// logits + preds (argmax stand-in for the JAX categorical sample)
// ---------------------------------------------------------------------------
__global__ void __launch_bounds__(1024)
logits_kernel(const float* __restrict__ h2, const float* __restrict__ context,
              const float* __restrict__ Wc, const float* __restrict__ bc,
              float* __restrict__ out, int step) {
  __shared__ float lg[BATCH][CLS];
  const int tid = threadIdx.x;
  for (int idx = tid; idx < BATCH * CLS; idx += 1024) {
    const int b = idx >> 6, n = idx & 63;
    const float* w = Wc + (size_t)n * (HID + VAL);
    float acc = bc[n];
    const float* hp = h2 + (size_t)b * HID;
    for (int j = 0; j < HID; j += 4) {
      v4f hh = *(const v4f*)(hp + j);
      v4f ww = *(const v4f*)(w + j);
      acc += hh[0] * ww[0] + hh[1] * ww[1] + hh[2] * ww[2] + hh[3] * ww[3];
    }
    const float* cp = context + (size_t)b * VAL;
    const float* w2 = w + HID;
    for (int j = 0; j < VAL; j += 4) {
      v4f cc = *(const v4f*)(cp + j);
      v4f ww = *(const v4f*)(w2 + j);
      acc += cc[0] * ww[0] + cc[1] * ww[1] + cc[2] * ww[2] + cc[3] * ww[3];
    }
    lg[b][n] = acc;
    out[((size_t)step * BATCH + b) * CLS + n] = acc;
  }
  __syncthreads();
  if (tid < BATCH) {
    int best = 0;
    float bv = lg[tid][0];
    for (int n = 1; n < CLS; ++n)
      if (lg[tid][n] > bv) { bv = lg[tid][n]; best = n; }
    out[(size_t)TSTEPS * BATCH * CLS + tid * TSTEPS + step] = (float)best;
  }
}

// ---------------------------------------------------------------------------
// host orchestration
// ---------------------------------------------------------------------------
extern "C" void kernel_launch(void* const* d_in, const int* in_sizes, int n_in,
                              void* d_out, int out_size, void* d_ws, size_t ws_size,
                              hipStream_t stream) {
  (void)in_sizes; (void)n_in; (void)out_size; (void)ws_size;
  // input order per setup_inputs()
  const float* listener = (const float*)d_in[0];
  const int*   lens     = (const int*)d_in[1];
  const int*   targets  = (const int*)d_in[2];
  // d_in[3] teacher_forcing == 1 (coin always true -> tokens from targets)
  const float* embed    = (const float*)d_in[4];
  const float* W_ih1    = (const float*)d_in[5];
  const float* W_hh1    = (const float*)d_in[6];
  const float* b_ih1    = (const float*)d_in[7];
  const float* b_hh1    = (const float*)d_in[8];
  const float* W_ih2    = (const float*)d_in[9];
  const float* W_hh2    = (const float*)d_in[10];
  const float* b_ih2    = (const float*)d_in[11];
  const float* b_hh2    = (const float*)d_in[12];
  const float* W_s      = (const float*)d_in[13];
  const float* b_s      = (const float*)d_in[14];
  const float* W_h      = (const float*)d_in[15];
  const float* b_h      = (const float*)d_in[16];
  const float* W_v      = (const float*)d_in[17];
  const float* b_v      = (const float*)d_in[18];
  const float* W_c      = (const float*)d_in[19];
  const float* b_c      = (const float*)d_in[20];
  const float* h1_0     = (const float*)d_in[21];
  const float* c1_0     = (const float*)d_in[22];
  const float* h2_0     = (const float*)d_in[23];
  const float* c2_0     = (const float*)d_in[24];

  // workspace carve-up (256B aligned)
  char* p = (char*)d_ws;
  auto carve = [&](size_t bytes) -> void* {
    void* r = (void*)p;
    p += (bytes + 255) & ~(size_t)255;
    return r;
  };
  float*  key_enc = (float*)carve((size_t)BATCH * TENC * KEY * 4);
  float*  value   = (float*)carve((size_t)BATCH * TENC * VAL * 4);
  __bf16* Wih1b   = (__bf16*)carve((size_t)GATES * XDIM * 2);
  __bf16* Whh1b   = (__bf16*)carve((size_t)GATES * HID * 2);
  __bf16* Wih2b   = (__bf16*)carve((size_t)GATES * HID * 2);
  __bf16* Whh2b   = (__bf16*)carve((size_t)GATES * HID * 2);
  float*  g1      = (float*)carve((size_t)BATCH * GATES * 4);
  float*  g2      = (float*)carve((size_t)BATCH * GATES * 4);
  float*  h1      = (float*)carve((size_t)BATCH * HID * 4);
  float*  c1      = (float*)carve((size_t)BATCH * HID * 4);
  float*  h2      = (float*)carve((size_t)BATCH * HID * 4);
  float*  c2      = (float*)carve((size_t)BATCH * HID * 4);
  float*  ctx     = (float*)carve((size_t)BATCH * VAL * 4);
  float*  q       = (float*)carve((size_t)BATCH * KEY * 4);
  float*  xbuf    = (float*)carve((size_t)BATCH * XDIM * 4);

  float* out      = (float*)d_out;
  float* atts_out = out + (size_t)TSTEPS * BATCH * CLS + (size_t)BATCH * TSTEPS;

  // 1) convert gate weights to bf16 (once; L2-resident afterwards)
  {
    int n1 = GATES * XDIM, n2 = GATES * HID;
    cvt_bf16_kernel<<<(n1 + 255) / 256, 256, 0, stream>>>(W_ih1, Wih1b, n1);
    cvt_bf16_kernel<<<(n2 + 255) / 256, 256, 0, stream>>>(W_hh1, Whh1b, n2);
    cvt_bf16_kernel<<<(n2 + 255) / 256, 256, 0, stream>>>(W_ih2, Wih2b, n2);
    cvt_bf16_kernel<<<(n2 + 255) / 256, 256, 0, stream>>>(W_hh2, Whh2b, n2);
  }

  // 2) key/value projections (WMMA)
  proj_kv_kernel<<<(BATCH * TENC) / 16, 256, 0, stream>>>(listener, W_h, b_h, W_v, b_v,
                                                          lens, key_enc, value);

  // 3) initial state
  {
    int ns = BATCH * HID;
    copy_f32_kernel<<<(ns + 255) / 256, 256, 0, stream>>>(h1_0, h1, ns);
    copy_f32_kernel<<<(ns + 255) / 256, 256, 0, stream>>>(c1_0, c1, ns);
    copy_f32_kernel<<<(ns + 255) / 256, 256, 0, stream>>>(h2_0, h2, ns);
    copy_f32_kernel<<<(ns + 255) / 256, 256, 0, stream>>>(c2_0, c2, ns);
    zero_f32_kernel<<<(BATCH * VAL + 255) / 256, 256, 0, stream>>>(ctx, BATCH * VAL);
  }

  // 4) sequential decode
  for (int step = 0; step < TSTEPS; ++step) {
    if (step > 0) {  // step 0 keeps the learned initial states
      build_x_kernel<<<1, 1024, 0, stream>>>(targets, step, embed, ctx, xbuf);
      gates_kernel<<<16, 256, 0, stream>>>(xbuf, XDIM, Wih1b, h1, HID, Whh1b,
                                           b_ih1, b_hh1, g1);
      lstm_update_kernel<<<16, 1024, 0, stream>>>(g1, h1, c1);
      gates_kernel<<<16, 256, 0, stream>>>(h1, HID, Wih2b, h2, HID, Whh2b,
                                           b_ih2, b_hh2, g2);
      lstm_update_kernel<<<16, 1024, 0, stream>>>(g2, h2, c2);
    }
    query_kernel<<<1, 1024, 0, stream>>>(h2, W_s, b_s, q);
    attention_kernel<<<BATCH, 1024, 0, stream>>>(q, key_enc, value, lens, ctx,
                                                 atts_out, step);
    logits_kernel<<<1, 1024, 0, stream>>>(h2, ctx, W_c, b_c, out, step);
  }
}
